// TransformerEncoderLevel_37391985279388
// MI455X (gfx1250) — matrix operand order repaired and checked by reading, not compiled, not run on hardware
//
#include <hip/hip_runtime.h>
#include <hip/hip_bf16.h>

typedef __bf16 bf16;
typedef __attribute__((ext_vector_type(16))) __bf16 v16bf;
typedef __attribute__((ext_vector_type(8)))  __bf16 bf16x8;
typedef __attribute__((ext_vector_type(8)))  float  v8f;

#define B_   4
#define S_   1024
#define D_   1024
#define H_   16
#define F_   4096
#define MBLK 128
#define NB_  8
#define DH_  64

// ---------------------------------------------------------------------------
// gfx1250 async global->LDS copy (ASYNCcnt-tracked); 16B per lane.
// VDST = VGPR with LDS byte offset, VADDR = 64-bit global address.
// Generic pointers to __shared__ carry the LDS offset in their low 32 bits.
// ---------------------------------------------------------------------------
__device__ __forceinline__ void async_copy_b128(unsigned lds_off,
                                                const void* gptr) {
  asm volatile("global_load_async_to_lds_b128 %0, %1, off"
               :: "v"(lds_off), "v"(gptr) : "memory");
}
__device__ __forceinline__ void wait_async0() {
  asm volatile("s_wait_asynccnt 0x0" ::: "memory");
}
__device__ __forceinline__ void wait_async4() {
  asm volatile("s_wait_asynccnt 0x4" ::: "memory");
}

// ---------------------------------------------------------------------------
// WMMA fragment loaders (CDNA5 ISA 7.12.2 layouts, wave32)
// A (16x32 bf16): lane r=L&15 -> row; lanes 0-15: K=0..7 & 16..23, lanes 16-31:
//   K=8..15 & 24..31  -> two b128 reads at +lh*8 and +16+lh*8
// B (32x16 bf16): lane r=L&15 -> col N; lanes 0-15: K=0..7 & 16..23, lanes 16-31:
//   K=8..15 & 24..31  -> two b128 reads at +lh*8 and +16+lh*8
// ---------------------------------------------------------------------------
__device__ __forceinline__ v16bf load_a_frag(const bf16* rowp, int lh) {
  bf16x8 lo = *(const bf16x8*)(rowp + lh * 8);
  bf16x8 hi = *(const bf16x8*)(rowp + 16 + lh * 8);
  v16bf r;
#pragma unroll
  for (int i = 0; i < 8; ++i) { r[i] = lo[i]; r[i + 8] = hi[i]; }
  return r;
}
__device__ __forceinline__ v16bf load_b_frag(const bf16* rowp, int lh) {
  bf16x8 lo = *(const bf16x8*)(rowp + lh * 8);
  bf16x8 hi = *(const bf16x8*)(rowp + 16 + lh * 8);
  v16bf r;
#pragma unroll
  for (int i = 0; i < 8; ++i) { r[i] = lo[i]; r[i + 8] = hi[i]; }
  return r;
}

// ---------------------------------------------------------------------------
// Conversion kernels
// ---------------------------------------------------------------------------
__global__ __launch_bounds__(256) void cvt_kernel(const float* __restrict__ in,
                                                  bf16* __restrict__ out, int n) {
  int i = blockIdx.x * 256 + threadIdx.x;
  if (i < n) out[i] = (bf16)in[i];
}

// in [Kd][Nd] row-major -> out [Nd][Kd] row-major (transposed), bf16
__global__ __launch_bounds__(256) void cvtT_kernel(const float* __restrict__ in,
                                                   bf16* __restrict__ out,
                                                   int Kd, int Nd) {
  int i = blockIdx.x * 256 + threadIdx.x;
  if (i < Kd * Nd) {
    int nn = i / Kd, kk = i - nn * Kd;
    out[i] = (bf16)in[(long)kk * Nd + nn];
  }
}

// ---------------------------------------------------------------------------
// Tiled bf16 WMMA GEMM: C[M,N] = A[M,K] @ Bt[N,K]^T (+bias, relu)
// 256 threads = 8 waves; block tile 128x128, K-step 32; double-buffered LDS
// filled with async global->LDS copies (ASYNCcnt pipelined, last iter peeled).
// Wave (wr,wc): rows wr*64..+64 (4 tiles), cols wc*32..+32 (2 tiles).
// ---------------------------------------------------------------------------
#define BM 128
#define BN 128
#define BK 32
#define PA 40   // LDS pitch (bf16): 80B rows, 16B-aligned chunks

// straight-line stage of one 128x32 A tile + 128x32 B tile (4 async per wave)
__device__ __forceinline__ void stage_tile(unsigned asoff, unsigned bsoff,
                                           const bf16* A, const bf16* Bt,
                                           long bm, long bn, int K, int k0,
                                           int row0, int cc0) {
  unsigned l0 = (unsigned)(row0 * PA + cc0 * 8) * 2;
  unsigned l1 = (unsigned)((row0 + 64) * PA + cc0 * 8) * 2;
  async_copy_b128(asoff + l0, &A[(bm + row0) * K + k0 + cc0 * 8]);
  async_copy_b128(bsoff + l0, &Bt[(bn + row0) * K + k0 + cc0 * 8]);
  async_copy_b128(asoff + l1, &A[(bm + row0 + 64) * K + k0 + cc0 * 8]);
  async_copy_b128(bsoff + l1, &Bt[(bn + row0 + 64) * K + k0 + cc0 * 8]);
}

__device__ __forceinline__ void gemm_compute(const bf16* Ac, const bf16* Bc,
                                             v8f acc[4][2], int wr, int wc,
                                             int lr, int lh) {
  v16bf af[4], bfg[2];
#pragma unroll
  for (int tr = 0; tr < 4; ++tr)
    af[tr] = load_a_frag(&Ac[(wr * 64 + tr * 16 + lr) * PA], lh);
#pragma unroll
  for (int tc = 0; tc < 2; ++tc)
    bfg[tc] = load_b_frag(&Bc[(wc * 32 + tc * 16 + lr) * PA], lh);
#pragma unroll
  for (int tr = 0; tr < 4; ++tr)
#pragma unroll
    for (int tc = 0; tc < 2; ++tc)
      acc[tr][tc] = __builtin_amdgcn_wmma_f32_16x16x32_bf16(
          false, af[tr], false, bfg[tc], (short)0, acc[tr][tc], false, false);
}

__global__ __launch_bounds__(256) void gemm_bf16(
    const bf16* __restrict__ A, const bf16* __restrict__ Bt,
    const float* __restrict__ bias, int relu,
    float* __restrict__ outF, bf16* __restrict__ outB,
    int M, int N, int K) {
  __shared__ __align__(16) bf16 As[2][BM * PA];
  __shared__ __align__(16) bf16 Bs[2][BN * PA];
  const int tid = threadIdx.x;
  const int lane = tid & 31, w = tid >> 5;
  const int lh = lane >> 4, lr = lane & 15;
  const int wr = w >> 2, wc = w & 3;
  const long bm = (long)blockIdx.y * BM, bn = (long)blockIdx.x * BN;
  const int row0 = tid >> 2, cc0 = tid & 3;

  unsigned asb[2] = {(unsigned)(size_t)&As[0][0], (unsigned)(size_t)&As[1][0]};
  unsigned bsb[2] = {(unsigned)(size_t)&Bs[0][0], (unsigned)(size_t)&Bs[1][0]};

  v8f acc[4][2];
#pragma unroll
  for (int i = 0; i < 4; ++i)
#pragma unroll
    for (int j = 0; j < 2; ++j)
      acc[i][j] = (v8f){0.f, 0.f, 0.f, 0.f, 0.f, 0.f, 0.f, 0.f};

  stage_tile(asb[0], bsb[0], A, Bt, bm, bn, K, 0, row0, cc0);

  const int nIter = K >> 5;
  for (int it = 0; it < nIter - 1; ++it) {
    const int cur = it & 1;
    stage_tile(asb[cur ^ 1], bsb[cur ^ 1], A, Bt, bm, bn, K, (it + 1) << 5,
               row0, cc0);
    wait_async4();  // async loads complete in order: current buffer landed
    __syncthreads();
    gemm_compute(&As[cur][0], &Bs[cur][0], acc, wr, wc, lr, lh);
    __syncthreads();
  }
  wait_async0();
  __syncthreads();
  gemm_compute(&As[(nIter - 1) & 1][0], &Bs[(nIter - 1) & 1][0], acc, wr, wc,
               lr, lh);

  // Epilogue. C layout: VGPR rr, lane L -> row rr+8*(L>>4), col L&15.
#pragma unroll
  for (int tr = 0; tr < 4; ++tr) {
#pragma unroll
    for (int tc = 0; tc < 2; ++tc) {
      int gn = (int)bn + wc * 32 + tc * 16 + lr;
      float bv = bias ? bias[gn] : 0.f;
#pragma unroll
      for (int rr = 0; rr < 8; ++rr) {
        long gm = bm + wr * 64 + tr * 16 + rr + 8 * lh;
        float v = acc[tr][tc][rr] + bv;
        if (relu) v = fmaxf(v, 0.f);
        long idx = gm * N + gn;
        if (outF) outF[idx] = v;
        if (outB) outB[idx] = (bf16)v;
      }
    }
  }
}

// ---------------------------------------------------------------------------
// Block-diagonal windowed attention. One workgroup per (nb, h, b).
// Phase 1: scores = Q@K^T/8 with mask; register softmax (half-wave shuffles).
// Phase 2: attn(bf16, LDS) @ V(staged transposed) -> AO.
// ---------------------------------------------------------------------------
#define PQ 72    // Q/K LDS pitch (bf16)
#define PT 136   // attn / V^T LDS pitch (bf16)

__global__ __launch_bounds__(256) void attn_kernel(
    const bf16* __restrict__ Q, const bf16* __restrict__ Km,
    const bf16* __restrict__ V, const int* __restrict__ vlen,
    bf16* __restrict__ AO) {
  __shared__ __align__(16) char smem[18432 + 34816];
  bf16* QS  = (bf16*)smem;               // [128][PQ] phase 1
  bf16* KS  = (bf16*)(smem + 18432);     // [128][PQ] phase 1
  bf16* VS  = (bf16*)smem;               // [64][PT]  phase 2 (over QS)
  bf16* ATT = (bf16*)(smem + 18432);     // [128][PT] phase 2 (over KS)

  const int tid = threadIdx.x;
  const int lane = tid & 31, w = tid >> 5;
  const int lh = lane >> 4, lr = lane & 15;
  const int nb = blockIdx.x, h = blockIdx.y, b = blockIdx.z;
  const long rowbase = (long)(b * S_ + nb * MBLK) * D_ + h * DH_;
  const int srow = tid >> 3, scc = tid & 7;  // staging: 32 rows/iter, 8 chunks

  // Stage Q, K blocks (128 x 64 bf16 each) via async global->LDS
  {
    unsigned qsb = (unsigned)(size_t)QS;
    unsigned ksb = (unsigned)(size_t)KS;
#pragma unroll
    for (int i = 0; i < 4; ++i) {
      int row = srow + 32 * i;
      long g = rowbase + (long)row * D_ + scc * 8;
      unsigned l = (unsigned)(row * PQ + scc * 8) * 2;
      async_copy_b128(qsb + l, &Q[g]);
      async_copy_b128(ksb + l, &Km[g]);
    }
  }
  wait_async0();
  __syncthreads();

  // scores: wave strip = 16 query rows x 128 keys = 8 tiles
  v8f s[8];
#pragma unroll
  for (int i = 0; i < 8; ++i)
    s[i] = (v8f){0.f, 0.f, 0.f, 0.f, 0.f, 0.f, 0.f, 0.f};
#pragma unroll
  for (int ks = 0; ks < 2; ++ks) {
    v16bf aq = load_a_frag(&QS[(w * 16 + lr) * PQ + ks * 32], lh);
#pragma unroll
    for (int tc = 0; tc < 8; ++tc) {
      v16bf bk = load_b_frag(&KS[(tc * 16 + lr) * PQ + ks * 32], lh);
      s[tc] = __builtin_amdgcn_wmma_f32_16x16x32_bf16(
          false, aq, false, bk, (short)0, s[tc], false, false);
    }
  }

  // scale + valid-length mask (reference: where(valid, s/8, -1e6))
  const int vl = vlen[b];
#pragma unroll
  for (int tc = 0; tc < 8; ++tc) {
    bool ok = (nb * MBLK + tc * 16 + lr) < vl;
#pragma unroll
    for (int rr = 0; rr < 8; ++rr) {
      float v = s[tc][rr] * 0.125f;
      s[tc][rr] = ok ? v : -1e6f;
    }
  }

  // softmax per row; element (tc,rr) belongs to row rr+8*lh; xor masks <16
  // keep reductions within the correct 16-lane half.
  float inv[8];
#pragma unroll
  for (int rr = 0; rr < 8; ++rr) {
    float mv = s[0][rr];
#pragma unroll
    for (int tc = 1; tc < 8; ++tc) mv = fmaxf(mv, s[tc][rr]);
    mv = fmaxf(mv, __shfl_xor(mv, 1, 32));
    mv = fmaxf(mv, __shfl_xor(mv, 2, 32));
    mv = fmaxf(mv, __shfl_xor(mv, 4, 32));
    mv = fmaxf(mv, __shfl_xor(mv, 8, 32));
    float sum = 0.f;
#pragma unroll
    for (int tc = 0; tc < 8; ++tc) {
      float e = __expf(s[tc][rr] - mv);
      s[tc][rr] = e;
      sum += e;
    }
    sum += __shfl_xor(sum, 1, 32);
    sum += __shfl_xor(sum, 2, 32);
    sum += __shfl_xor(sum, 4, 32);
    sum += __shfl_xor(sum, 8, 32);
    inv[rr] = 1.f / sum;
  }
  __syncthreads();  // QS/KS reads complete before overlay writes

  // write attn (bf16) strip; stage V transposed [dh][key]
#pragma unroll
  for (int tc = 0; tc < 8; ++tc)
#pragma unroll
    for (int rr = 0; rr < 8; ++rr) {
      int row = w * 16 + rr + 8 * lh;
      ATT[row * PT + tc * 16 + lr] = (bf16)(s[tc][rr] * inv[rr]);
    }
#pragma unroll
  for (int i = 0; i < 4; ++i) {
    int key = srow + 32 * i;
    bf16x8 vv = *(const bf16x8*)&V[rowbase + (long)key * D_ + scc * 8];
#pragma unroll
    for (int e = 0; e < 8; ++e) VS[(scc * 8 + e) * PT + key] = vv[e];
  }
  __syncthreads();

  // o = attn @ V : wave strip 16 queries x 64 dh = 4 tiles, K=128 keys
  v8f o[4];
#pragma unroll
  for (int i = 0; i < 4; ++i)
    o[i] = (v8f){0.f, 0.f, 0.f, 0.f, 0.f, 0.f, 0.f, 0.f};
#pragma unroll
  for (int ks = 0; ks < 4; ++ks) {
    v16bf aa = load_a_frag(&ATT[(w * 16 + lr) * PT + ks * 32], lh);
#pragma unroll
    for (int tc = 0; tc < 4; ++tc) {
      v16bf bv = load_b_frag(&VS[(tc * 16 + lr) * PT + ks * 32], lh);
      o[tc] = __builtin_amdgcn_wmma_f32_16x16x32_bf16(
          false, aa, false, bv, (short)0, o[tc], false, false);
    }
  }
#pragma unroll
  for (int tc = 0; tc < 4; ++tc)
#pragma unroll
    for (int rr = 0; rr < 8; ++rr) {
      int q = w * 16 + rr + 8 * lh;
      AO[rowbase + (long)q * D_ + tc * 16 + lr] = (bf16)o[tc][rr];
    }
}

// ---------------------------------------------------------------------------
// Fused residual-add + LayerNorm (one block per row of D=1024)
// ---------------------------------------------------------------------------
__device__ __forceinline__ float block_sum(float v, float* red, float* broad,
                                           int tid) {
#pragma unroll
  for (int off = 1; off < 32; off <<= 1) v += __shfl_xor(v, off, 32);
  if ((tid & 31) == 0) red[tid >> 5] = v;
  __syncthreads();
  if (tid == 0) {
    float t = 0.f;
#pragma unroll
    for (int i = 0; i < 8; ++i) t += red[i];
    *broad = t;
  }
  __syncthreads();
  float r = *broad;
  __syncthreads();  // safe reuse of red/broad
  return r;
}

__global__ __launch_bounds__(256) void addnorm_kernel(
    const float* __restrict__ X, const float* __restrict__ R,
    const float* __restrict__ g, const float* __restrict__ be,
    float* __restrict__ Yf, bf16* __restrict__ Yb) {
  __shared__ float red[8];
  __shared__ float broad;
  const int tid = threadIdx.x;
  const long base = (long)blockIdx.x * D_;
  float v[4];
  float sum = 0.f;
#pragma unroll
  for (int j = 0; j < 4; ++j) {
    int c = tid + j * 256;
    float t = X[base + c] + R[base + c];
    v[j] = t;
    sum += t;
  }
  float mean = block_sum(sum, red, &broad, tid) * (1.f / (float)D_);
  float s2 = 0.f;
#pragma unroll
  for (int j = 0; j < 4; ++j) {
    float d = v[j] - mean;
    s2 += d * d;
  }
  float var = block_sum(s2, red, &broad, tid) * (1.f / (float)D_);
  float is = rsqrtf(var + 1e-5f);
#pragma unroll
  for (int j = 0; j < 4; ++j) {
    int c = tid + j * 256;
    float ov = (v[j] - mean) * is * g[c] + be[c];
    Yf[base + c] = ov;
    if (Yb) Yb[base + c] = (bf16)ov;
  }
}

// ---------------------------------------------------------------------------
extern "C" void kernel_launch(void* const* d_in, const int* in_sizes, int n_in,
                              void* d_out, int out_size, void* d_ws,
                              size_t ws_size, hipStream_t stream) {
  const float* X   = (const float*)d_in[0];
  const int*   vl  = (const int*)d_in[1];
  const float* Wq  = (const float*)d_in[2];
  const float* Wk  = (const float*)d_in[3];
  const float* Wv  = (const float*)d_in[4];
  const float* Wo  = (const float*)d_in[5];
  const float* g1  = (const float*)d_in[6];
  const float* be1 = (const float*)d_in[7];
  const float* W1  = (const float*)d_in[8];
  const float* bf1 = (const float*)d_in[9];
  const float* W2  = (const float*)d_in[10];
  const float* bf2 = (const float*)d_in[11];
  const float* g2  = (const float*)d_in[12];
  const float* be2 = (const float*)d_in[13];

  char* ws = (char*)d_ws;
  const size_t MB1 = (size_t)1 << 20;
  bf16*  Xb  = (bf16*)(ws + 0 * MB1);   // 8MB
  bf16*  Qb  = (bf16*)(ws + 8 * MB1);   // 8MB
  bf16*  Kb  = (bf16*)(ws + 16 * MB1);  // 8MB
  bf16*  Vb  = (bf16*)(ws + 24 * MB1);  // 8MB
  bf16*  Hb  = (bf16*)(ws + 0 * MB1);   // 32MB, overlays Xb..Vb (dead by then)
  bf16*  Wqt = (bf16*)(ws + 32 * MB1);  // 2MB each
  bf16*  Wkt = (bf16*)(ws + 34 * MB1);
  bf16*  Wvt = (bf16*)(ws + 36 * MB1);
  bf16*  Wot = (bf16*)(ws + 38 * MB1);
  bf16*  W1t = (bf16*)(ws + 40 * MB1);  // 8MB
  bf16*  W2t = (bf16*)(ws + 48 * MB1);  // 8MB
  bf16*  AOb = (bf16*)(ws + 56 * MB1);  // 8MB
  float* Sf  = (float*)(ws + 64 * MB1); // 16MB (O-proj, then FFN2 out)
  float* Yf  = (float*)(ws + 80 * MB1); // 16MB
  bf16*  Yb  = (bf16*)(ws + 96 * MB1);  // 8MB

  const int T = 256;
  const int MR = B_ * S_;  // 4096 rows

  cvt_kernel<<<(MR * D_) / T, T, 0, stream>>>(X, Xb, MR * D_);
  cvtT_kernel<<<(D_ * D_) / T, T, 0, stream>>>(Wq, Wqt, D_, D_);
  cvtT_kernel<<<(D_ * D_) / T, T, 0, stream>>>(Wk, Wkt, D_, D_);
  cvtT_kernel<<<(D_ * D_) / T, T, 0, stream>>>(Wv, Wvt, D_, D_);
  cvtT_kernel<<<(D_ * D_) / T, T, 0, stream>>>(Wo, Wot, D_, D_);
  cvtT_kernel<<<(D_ * F_) / T, T, 0, stream>>>(W1, W1t, D_, F_);
  cvtT_kernel<<<(F_ * D_) / T, T, 0, stream>>>(W2, W2t, F_, D_);

  dim3 gQKV(D_ / BN, MR / BM);
  gemm_bf16<<<gQKV, T, 0, stream>>>(Xb, Wqt, nullptr, 0, nullptr, Qb, MR, D_, D_);
  gemm_bf16<<<gQKV, T, 0, stream>>>(Xb, Wkt, nullptr, 0, nullptr, Kb, MR, D_, D_);
  gemm_bf16<<<gQKV, T, 0, stream>>>(Xb, Wvt, nullptr, 0, nullptr, Vb, MR, D_, D_);

  attn_kernel<<<dim3(NB_, H_, B_), T, 0, stream>>>(Qb, Kb, Vb, vl, AOb);

  gemm_bf16<<<gQKV, T, 0, stream>>>(AOb, Wot, nullptr, 0, Sf, nullptr, MR, D_, D_);
  addnorm_kernel<<<MR, T, 0, stream>>>(X, Sf, g1, be1, Yf, Yb);

  gemm_bf16<<<dim3(F_ / BN, MR / BM), T, 0, stream>>>(Yb, W1t, bf1, 1, nullptr,
                                                      Hb, MR, F_, D_);
  gemm_bf16<<<dim3(D_ / BN, MR / BM), T, 0, stream>>>(Hb, W2t, bf2, 0, Sf,
                                                      nullptr, MR, D_, F_);
  addnorm_kernel<<<MR, T, 0, stream>>>(Yf, Sf, g2, be2, (float*)d_out, nullptr);
}
